// DeepIRT_67018669686876
// MI455X (gfx1250) — matrix operand order repaired and checked by reading, not compiled, not run on hardware
//
#include <hip/hip_runtime.h>
#include <hip/hip_bf16.h>

#define B_  4096
#define T_  50
#define K_  32
#define D_  64
#define H_  256
#define HL_ 128
#define NG  512   // 4*HL

typedef __bf16 bf16;
typedef __attribute__((ext_vector_type(16))) __bf16 v16bf;
typedef __attribute__((ext_vector_type(8)))  float  v8f;

__device__ __forceinline__ float sigmoidf_(float x){ return 1.0f/(1.0f+__expf(-x)); }
__device__ __forceinline__ float tanhfast_(float x){ return 2.0f/(1.0f+__expf(-2.0f*x)) - 1.0f; }

// Contiguous 32B fragment load (pre-swizzled LDS): 2x ds_load_b128
__device__ __forceinline__ v16bf load_frag(const bf16* base, int tile, int lane){
  return *(const v16bf*)(base + (tile*32 + lane)*16);
}

// ---------------- Kernel 1: gathers + mastery + theta/a DNNs ----------------
__global__ __launch_bounds__(256)
void gather_dnn_kernel(const int* uididx, const int* kcodeidx, const int* kcode_len,
                       const float* stuE, const float* knE,
                       const float* TW1, const float* Tb1, const float* TW2, const float* Tb2,
                       const float* AW1, const float* Ab1, const float* AW2, const float* Ab2,
                       float* theta_out, float* a_out)
{
  __shared__ float sK[K_*D_];
  __shared__ float sStu[D_];
  __shared__ float sMast[K_];
  __shared__ float sMV[D_];
  __shared__ float sAV[D_];
  __shared__ float sRed[256];
  const int b = blockIdx.x, tid = threadIdx.x;
  const int klen = kcode_len[b];
  if (tid < D_) sStu[tid] = stuE[uididx[b]*D_ + tid];
  for (int e = tid; e < K_*D_; e += 256) {
    int k = e / D_, d = e % D_;
    sK[e] = knE[kcodeidx[b*K_ + k]*D_ + d];
  }
  __syncthreads();
  if (tid < K_) {
    float s = 0.f;
    for (int d=0; d<D_; ++d) s += sK[tid*D_+d]*sStu[d];
    sMast[tid] = sigmoidf_(s*0.2f);
  }
  __syncthreads();
  if (tid < D_) {
    float mv=0.f, av=0.f;
    for (int k=0;k<klen;++k){ float e = sK[k*D_+tid]; av += e; mv += sMast[k]*e; }
    sMV[tid]=mv; sAV[tid]=av;
  }
  __syncthreads();
  {
    float acc = 0.f;
    for (int d=0; d<D_; ++d) acc += sMV[d]*TW1[d*H_ + tid];
    sRed[tid] = tanhf(acc + Tb1[tid]) * TW2[tid];
  }
  __syncthreads();
  for (int s = 128; s > 0; s >>= 1) { if (tid < s) sRed[tid] += sRed[tid+s]; __syncthreads(); }
  float tval = 0.f;
  if (tid == 0) tval = sRed[0] + Tb2[0];
  __syncthreads();
  {
    float acc = 0.f;
    for (int d=0; d<D_; ++d) acc += sAV[d]*AW1[d*H_ + tid];
    sRed[tid] = tanhf(acc + Ab1[tid]) * AW2[tid];
  }
  __syncthreads();
  for (int s = 128; s > 0; s >>= 1) { if (tid < s) sRed[tid] += sRed[tid+s]; __syncthreads(); }
  if (tid == 0) {
    theta_out[b] = tval;
    float av = sRed[0] + Ab2[0];
    a_out[b] = 8.f*(sigmoidf_(fabsf(av)) - 0.5f);
  }
}

// ---------------- Kernel 2: WMMA masked softmax attention -> bvec ----------------
// One block per batch element; 128 threads = 4 waves = 4 M-tiles (T=50 padded to 64).
// scores(64x32) = Q(64x64) @ kemb^T : per wave 4 WMMAs (2 n-tiles x 2 k-chunks)
// bvec(64x64)   = P(64x32) @ kemb   : per wave 4 WMMAs (4 n-tiles x 1 k-chunk)
__global__ __launch_bounds__(128)
void attn_kernel(const int* kcodeidx, const int* kcode_len, const float* qidemb,
                 const float* knE, float* bvec)
{
  __shared__ __attribute__((aligned(32))) bf16 sQA[4*2*32*16]; // Q A-frags   8 KB
  __shared__ __attribute__((aligned(32))) bf16 sKB[2*2*32*16]; // kemb^T B    4 KB
  __shared__ __attribute__((aligned(32))) bf16 sVB[4*32*16];   // kemb B      4 KB
  __shared__ __attribute__((aligned(32))) bf16 sPA[4*32*16];   // attw A      4 KB
  __shared__ float sS[64*32];                                  // scores      8 KB
  const int b = blockIdx.x, tid = threadIdx.x;
  const int lane = tid & 31, wave = tid >> 5;
  const int half = lane >> 4, nlo = lane & 15;
  const int klen = kcode_len[b];

  // stage Q as A-frags (rows >= T_ zero-padded)
  for (int e = tid; e < 64*64; e += 128) {
    int m = e >> 6, d = e & 63;
    float v = (m < T_) ? qidemb[(b*T_ + m)*D_ + d] : 0.f;
    int mt = m >> 4, ml = m & 15;
    int kc = d >> 5, k32 = d & 31;
    int ha = (k32 >> 3) & 1, j = (k32 & 7) | ((k32 & 16) ? 8 : 0);
    sQA[((mt*2+kc)*32 + ha*16 + ml)*16 + j] = (bf16)v;
  }
  // stage kemb into both B layouts
  for (int e = tid; e < K_*D_; e += 128) {
    int k = e >> 6, d = e & 63;                       // kemb[k][d]
    float v = knE[kcodeidx[b*K_+k]*D_+d];
    { // scores B: K-dim = d (2 chunks), N = k (2 tiles)
      int kc = d >> 5, k32 = d & 31;
      int khalf = (k32 >> 3) & 1, j = (k32 & 7) | ((k32 & 16) ? 8 : 0);
      int nt = k >> 4, laneB = khalf*16 + (k & 15);
      sKB[((kc*2+nt)*32 + laneB)*16 + j] = (bf16)v;
    }
    { // bvec B: K-dim = k (1 chunk), N = d (4 tiles)
      int khalf = (k >> 3) & 1, j = (k & 7) | ((k & 16) ? 8 : 0);
      int nt = d >> 4, laneB = khalf*16 + (d & 15);
      sVB[(nt*32 + laneB)*16 + j] = (bf16)v;
    }
  }
  __syncthreads();

  // scores for M-tile = wave
  {
    v16bf a0 = *(const v16bf*)(sQA + (wave*2+0)*512 + lane*16);
    v16bf a1 = *(const v16bf*)(sQA + (wave*2+1)*512 + lane*16);
    v8f acc0 = {}, acc1 = {};
    acc0 = __builtin_amdgcn_wmma_f32_16x16x32_bf16(false, a0, false, load_frag(sKB, 0, lane), (short)0, acc0, false, false);
    acc0 = __builtin_amdgcn_wmma_f32_16x16x32_bf16(false, a1, false, load_frag(sKB, 2, lane), (short)0, acc0, false, false);
    acc1 = __builtin_amdgcn_wmma_f32_16x16x32_bf16(false, a0, false, load_frag(sKB, 1, lane), (short)0, acc1, false, false);
    acc1 = __builtin_amdgcn_wmma_f32_16x16x32_bf16(false, a1, false, load_frag(sKB, 3, lane), (short)0, acc1, false, false);
#pragma unroll
    for (int r = 0; r < 8; ++r) {
      int m = wave*16 + r + 8*half;
      sS[m*32 + nlo]      = (nlo      < klen) ? acc0[r]*0.15f : -1e9f;
      sS[m*32 + 16 + nlo] = (16 + nlo < klen) ? acc1[r]*0.15f : -1e9f;
    }
  }
  __syncthreads();

  // row softmax (threads 0..63 own one row; pad rows produce zero P)
  if (tid < 64) {
    int m = tid;
    float p[K_]; float mx = -1e30f;
#pragma unroll
    for (int k=0;k<K_;++k){ float v = sS[m*32+k]; p[k]=v; mx=fmaxf(mx,v); }
    float sum=0.f;
#pragma unroll
    for (int k=0;k<K_;++k){ float e=__expf(p[k]-mx); p[k]=e; sum+=e; }
    float inv = (m < T_) ? 1.f/sum : 0.f;
    int mt = m >> 4, ml = m & 15;
#pragma unroll
    for (int k=0;k<K_;++k){
      int ha=(k>>3)&1, j=(k&7)|((k&16)?8:0);
      sPA[(mt*32 + ha*16 + ml)*16 + j] = (bf16)(p[k]*inv);
    }
  }
  __syncthreads();

  // bvec = P @ kemb for M-tile = wave
  {
    v16bf ap = *(const v16bf*)(sPA + wave*512 + lane*16);
#pragma unroll
    for (int nt = 0; nt < 4; ++nt) {
      v8f acc = {};
      acc = __builtin_amdgcn_wmma_f32_16x16x32_bf16(false, ap, false, load_frag(sVB, nt, lane), (short)0, acc, false, false);
#pragma unroll
      for (int r = 0; r < 8; ++r) {
        int m = wave*16 + r + 8*half;
        if (m < T_) bvec[(b*T_ + m)*D_ + nt*16 + nlo] = acc[r];
      }
    }
  }
}

// ---------------- Kernel 3: fused WMMA LSTM (xg + recurrence) ----------------
// Block: 256 threads (8 waves), 16 batch rows. Wave w computes i,f,g,o tiles for
// hidden slice [16w,16w+16) (tile column base = gate*128 + w*16) -> cell update in
// registers. All 24 loop-invariant B fragments (Wi/Wh) are hoisted into VGPRs; the
// t-loop does only 6 A-fragment loads + 24 WMMAs + masked h stores per wave.
__global__ __launch_bounds__(256)
void lstm_kernel(const float* bvec, const int* qid_len,
                 const float* Wi, const float* Wh, const float* Lb,
                 float* hT)
{
  __shared__ __attribute__((aligned(32))) bf16 sWhS[4*32*32*16];  // 128 KB
  __shared__ __attribute__((aligned(32))) bf16 sWiS[2*32*32*16];  //  64 KB
  __shared__ __attribute__((aligned(32))) bf16 sHA[4*32*16];      //   4 KB
  __shared__ __attribute__((aligned(32))) bf16 sXA[2][2*32*16];   //   4 KB (double buffer)
  __shared__ int sLen[16];

  const int tid    = threadIdx.x;
  const int lane   = tid & 31;
  const int wave   = tid >> 5;
  const int nlo    = lane & 15;
  const int myhalf = lane >> 4;
  const int row0   = blockIdx.x * 16;
  const int hc     = wave*16 + nlo;

  // ---- stage weights pre-swizzled into B-fragment order ----
  for (int e = tid; e < HL_*NG; e += 256) {
    int k = e >> 9, n = e & 511;
    int kc = k >> 5, k32 = k & 31;
    int khalf = (k32 >> 3) & 1, j = (k32 & 7) | ((k32 & 16) ? 8 : 0);
    int ntg = n >> 4, laneB = khalf*16 + (n & 15);
    sWhS[((kc*32 + ntg)*32 + laneB)*16 + j] = (bf16)Wh[e];
  }
  for (int e = tid; e < D_*NG; e += 256) {
    int k = e >> 9, n = e & 511;
    int kc = k >> 5, k32 = k & 31;
    int khalf = (k32 >> 3) & 1, j = (k32 & 7) | ((k32 & 16) ? 8 : 0);
    int ntg = n >> 4, laneB = khalf*16 + (n & 15);
    sWiS[((kc*32 + ntg)*32 + laneB)*16 + j] = (bf16)Wi[e];
  }
  for (int e = tid; e < 4*32*16; e += 256) sHA[e] = (bf16)0.f;
  // stage x tile for t=0
#pragma unroll
  for (int i = 0; i < 4; ++i) {
    int e = tid + i*256;
    int m = e >> 6, d = e & 63;
    int kc = d >> 5, k32 = d & 31;
    int ha = (k32 >> 3) & 1;
    int j  = (k32 & 7) | ((k32 & 16) ? 8 : 0);
    sXA[0][kc*512 + (ha*16 + m)*16 + j] = (bf16)bvec[((row0+m)*T_ + 0)*D_ + d];
  }
  if (tid < 16) sLen[tid] = qid_len[row0 + tid];
  __syncthreads();

  // ---- hoist loop-invariant B fragments into registers (24 frags) ----
  v16bf bwi0[4], bwi1[4], bwh0[4], bwh1[4], bwh2[4], bwh3[4];
#pragma unroll
  for (int g = 0; g < 4; ++g) {
    const int ntg = g*8 + wave;
    bwi0[g] = load_frag(sWiS, 0*32 + ntg, lane);
    bwi1[g] = load_frag(sWiS, 1*32 + ntg, lane);
    bwh0[g] = load_frag(sWhS, 0*32 + ntg, lane);
    bwh1[g] = load_frag(sWhS, 1*32 + ntg, lane);
    bwh2[g] = load_frag(sWhS, 2*32 + ntg, lane);
    bwh3[g] = load_frag(sWhS, 3*32 + ntg, lane);
  }

  // ---- per-thread loop-invariant state ----
  float bi  = Lb[       hc];
  float bfg = Lb[128  + hc];
  float bgg = Lb[256  + hc];
  float bog = Lb[384  + hc];
  const int kc_h  = hc >> 5;
  const int k32h  = hc & 31;
  const int halfA = (k32h >> 3) & 1;
  const int jA    = (k32h & 7) | ((k32h & 16) ? 8 : 0);
  const int hoff0 = kc_h*512 + halfA*256 + jA + 128*myhalf;   // + r*16
  int lenr[8];
#pragma unroll
  for (int r = 0; r < 8; ++r) lenr[r] = sLen[r + 8*myhalf];
  float cst[8];
#pragma unroll
  for (int r = 0; r < 8; ++r) cst[r] = 0.f;

  for (int t = 0; t < T_; ++t) {
    const bf16* xa = sXA[t & 1];
    v16bf ax0 = *(const v16bf*)(xa +   0 + lane*16);
    v16bf ax1 = *(const v16bf*)(xa + 512 + lane*16);
    v16bf ah0 = *(const v16bf*)(sHA +    0 + lane*16);
    v16bf ah1 = *(const v16bf*)(sHA +  512 + lane*16);
    v16bf ah2 = *(const v16bf*)(sHA + 1024 + lane*16);
    v16bf ah3 = *(const v16bf*)(sHA + 1536 + lane*16);

    v8f acc[4];
#pragma unroll
    for (int g = 0; g < 4; ++g) {
      v8f a = {};
      a = __builtin_amdgcn_wmma_f32_16x16x32_bf16(false, ax0, false, bwi0[g], (short)0, a, false, false);
      a = __builtin_amdgcn_wmma_f32_16x16x32_bf16(false, ax1, false, bwi1[g], (short)0, a, false, false);
      a = __builtin_amdgcn_wmma_f32_16x16x32_bf16(false, ah0, false, bwh0[g], (short)0, a, false, false);
      a = __builtin_amdgcn_wmma_f32_16x16x32_bf16(false, ah1, false, bwh1[g], (short)0, a, false, false);
      a = __builtin_amdgcn_wmma_f32_16x16x32_bf16(false, ah2, false, bwh2[g], (short)0, a, false, false);
      a = __builtin_amdgcn_wmma_f32_16x16x32_bf16(false, ah3, false, bwh3[g], (short)0, a, false, false);
      acc[g] = a;
    }
    __syncthreads();   // all waves done reading sHA / sXA[t&1]

    // LSTM cell update in registers; only h goes back to LDS (masked)
#pragma unroll
    for (int r = 0; r < 8; ++r) {
      float gi = acc[0][r] + bi;
      float gf = acc[1][r] + bfg;
      float gg = acc[2][r] + bgg;
      float go = acc[3][r] + bog;
      float c2 = sigmoidf_(gf)*cst[r] + sigmoidf_(gi)*tanhfast_(gg);
      float h2 = sigmoidf_(go)*tanhfast_(c2);
      if (t < lenr[r]) {
        cst[r] = c2;
        sHA[hoff0 + r*16] = (bf16)h2;
      }
    }
    // stage next x tile into the other buffer
    if (t + 1 < T_) {
      bf16* xw = sXA[(t + 1) & 1];
#pragma unroll
      for (int i = 0; i < 4; ++i) {
        int e = tid + i*256;
        int m = e >> 6, d = e & 63;
        int kc = d >> 5, k32 = d & 31;
        int ha = (k32 >> 3) & 1;
        int j  = (k32 & 7) | ((k32 & 16) ? 8 : 0);
        xw[kc*512 + (ha*16 + m)*16 + j] = (bf16)bvec[((row0+m)*T_ + (t+1))*D_ + d];
      }
    }
    __syncthreads();
  }

  // write hT from swizzled sHA
  for (int e = tid; e < 16*HL_; e += 256) {
    int m = e >> 7, h = e & 127;
    int kc = h >> 5, k32 = h & 31;
    int ha = (k32 >> 3) & 1;
    int j  = (k32 & 7) | ((k32 & 16) ? 8 : 0);
    hT[(row0+m)*HL_ + h] = (float)sHA[kc*512 + (ha*16 + m)*16 + j];
  }
}

// ---------------- Kernel 4: output head + IRT combine ----------------
__global__ __launch_bounds__(256)
void final_kernel(const float* hT, const float* Wo, const float* bo,
                  const float* theta, const float* aval, float* out)
{
  int b = blockIdx.x*256 + threadIdx.x;
  if (b >= B_) return;
  float acc = 0.f;
  for (int j=0;j<HL_;++j) acc += hT[b*HL_+j]*Wo[j];
  float bb = 8.f*(sigmoidf_(acc + bo[0]) - 0.5f);
  out[b] = sigmoidf_(aval[b]*(theta[b] - bb));
}

extern "C" void kernel_launch(void* const* d_in, const int* in_sizes, int n_in,
                              void* d_out, int out_size, void* d_ws, size_t ws_size,
                              hipStream_t stream) {
  const int*   uididx    = (const int*)  d_in[0];
  const int*   kcodeidx  = (const int*)  d_in[1];
  const int*   kcode_len = (const int*)  d_in[2];
  const float* qidemb    = (const float*)d_in[3];
  const int*   qid_len   = (const int*)  d_in[4];
  const float* stuE      = (const float*)d_in[5];
  const float* knE       = (const float*)d_in[6];
  const float* TW1 = (const float*)d_in[7];  const float* Tb1 = (const float*)d_in[8];
  const float* TW2 = (const float*)d_in[9];  const float* Tb2 = (const float*)d_in[10];
  const float* AW1 = (const float*)d_in[11]; const float* Ab1 = (const float*)d_in[12];
  const float* AW2 = (const float*)d_in[13]; const float* Ab2 = (const float*)d_in[14];
  const float* LWi = (const float*)d_in[15]; const float* LWh = (const float*)d_in[16];
  const float* Lb  = (const float*)d_in[17]; const float* LWo = (const float*)d_in[18];
  const float* Lbo = (const float*)d_in[19];

  float* ws    = (float*)d_ws;
  float* theta = ws;                       // [B]
  float* aval  = ws + B_;                  // [B]
  float* hTbuf = ws + 2*B_;                // [B, HL]
  float* bvec  = ws + 2*B_ + B_*HL_;       // [B, T, D]  (~52.4 MB)

  gather_dnn_kernel<<<B_, 256, 0, stream>>>(uididx, kcodeidx, kcode_len, stuE, knE,
                                            TW1, Tb1, TW2, Tb2, AW1, Ab1, AW2, Ab2,
                                            theta, aval);
  attn_kernel<<<B_, 128, 0, stream>>>(kcodeidx, kcode_len, qidemb, knE, bvec);
  lstm_kernel<<<B_/16, 256, 0, stream>>>(bvec, qid_len, LWi, LWh, Lb, hTbuf);
  final_kernel<<<B_/256, 256, 0, stream>>>(hTbuf, LWo, Lbo, theta, aval, (float*)d_out);
}
